// Attention_79267916415174
// MI455X (gfx1250) — matrix operand order repaired and checked by reading, not compiled, not run on hardware
//
#include <hip/hip_runtime.h>
#include <hip/hip_bf16.h>
#include <math.h>

// ---------------------------------------------------------------------------
// CDNA5 / gfx1250 wave32 WMMA attention-decoder step.
// All big GEMMs use v_wmma_f32_16x16x32_bf16 (f32 accumulate).
// ---------------------------------------------------------------------------

typedef __attribute__((ext_vector_type(16))) __bf16 bf16x16;
typedef __attribute__((ext_vector_type(8)))  float  f32x8;

#define WMMA_BF16(A, B, C) \
  __builtin_amdgcn_wmma_f32_16x16x32_bf16(false, (A), false, (B), (short)0, (C), false, false)

__device__ __forceinline__ float sigmoidf_(float x) { return 1.f / (1.f + expf(-x)); }

// A fragment (16x32 bf16): 16 contiguous K values, k = 16*half + j.
__device__ __forceinline__ bf16x16 load_a_bf16(const __bf16* __restrict__ row,
                                               int kb, int half) {
  bf16x16 af;
#pragma unroll
  for (int j = 0; j < 8; ++j) {
    af[j]     = row[kb + half * 16 + j];
    af[8 + j] = row[kb + half * 16 + 8 + j];
  }
  return af;
}

// A fragment from f32 memory with convert.
__device__ __forceinline__ bf16x16 load_a_f32cvt(const float* __restrict__ row,
                                                 int kb, int half) {
  bf16x16 af;
#pragma unroll
  for (int j = 0; j < 8; ++j) {
    af[j]     = (__bf16)row[kb + half * 16 + j];
    af[8 + j] = (__bf16)row[kb + half * 16 + 8 + j];
  }
  return af;
}

// B fragment: 16 contiguous K values at fixed N, converted f32->bf16.
__device__ __forceinline__ bf16x16 load_b_f32cvt(const float* __restrict__ p) {
  bf16x16 bf;
#pragma unroll
  for (int j = 0; j < 16; ++j) bf[j] = (__bf16)p[j];
  return bf;
}

// ---------------------------------------------------------------------------
// Kernel 1: fuse location conv weights with loc projection:
//   W2[a][j] = sum_f loc_w[a,f] * conv_w[f, c=j>>5, k=j&31]   (k==31 -> 0 pad)
//   b2[a]    = loc_b[a] + sum_f loc_w[a,f] * conv_b[f]
// ---------------------------------------------------------------------------
__global__ void k_prep(const float* __restrict__ conv_w, const float* __restrict__ conv_b,
                       const float* __restrict__ loc_w,  const float* __restrict__ loc_b,
                       __bf16* __restrict__ w2b, float* __restrict__ b2) {
  int a = threadIdx.x;
  if (a >= 128) return;
  float bb = loc_b[a];
  for (int f = 0; f < 32; ++f) bb += loc_w[a * 32 + f] * conv_b[f];
  b2[a] = bb;
  for (int j = 0; j < 64; ++j) {
    int c = j >> 5, kk = j & 31;
    float v = 0.f;
    if (kk < 31)
      for (int f = 0; f < 32; ++f)
        v += loc_w[a * 32 + f] * conv_w[f * 62 + c * 31 + kk];
    w2b[a * 64 + j] = (__bf16)v;
  }
}

// ---------------------------------------------------------------------------
// Kernel 2: pack X = concat(prenet[256], prev_context[512], h[1024]) -> bf16
// ---------------------------------------------------------------------------
__global__ void k_pack(const float* __restrict__ pre, const float* __restrict__ ctx,
                       const float* __restrict__ h, __bf16* __restrict__ Xb) {
  int k = blockIdx.x * 256 + threadIdx.x;   // 0..1791
  int b = blockIdx.y;
  float v;
  if (k < 256)       v = pre[b * 256 + k];
  else if (k < 768)  v = ctx[b * 512 + (k - 256)];
  else               v = h[b * 1024 + (k - 768)];
  Xb[(size_t)b * 1792 + k] = (__bf16)v;
}

// ---------------------------------------------------------------------------
// Kernel 3: gates[128,4096] = X @ [W_ih|W_hh]^T   (M=128,N=4096,K=1792)
// 1 wave = 1 N-tile x 4 M-tiles (B fragment reused 4x).  K loop split at the
// W_ih/W_hh boundary so all pointers are loop-invariant.
// grid = 64 blocks * 8 waves = 512 wave-tiles.
// ---------------------------------------------------------------------------
__device__ __forceinline__ void lstm_kstep(const __bf16* __restrict__ Xb,
                                           const float* __restrict__ wrow,
                                           int kb, int half, int ln, int mbase,
                                           f32x8* acc) {
  bf16x16 bf = load_b_f32cvt(wrow);
#pragma unroll
  for (int t = 0; t < 4; ++t) {
    bf16x16 af = load_a_bf16(Xb + (size_t)((mbase + t) * 16 + ln) * 1792, kb, half);
    acc[t] = WMMA_BF16(af, bf, acc[t]);
  }
}

__global__ __launch_bounds__(256) void k_lstm_gemm(
    const __bf16* __restrict__ Xb,
    const float* __restrict__ W_ih, const float* __restrict__ W_hh,
    float* __restrict__ gates) {
  int lane = threadIdx.x & 31, wave = threadIdx.x >> 5;
  int half = lane >> 4, ln = lane & 15, khalf = half * 16;
  int w  = blockIdx.x * 8 + wave;         // 0..511
  int nt = w >> 1;                        // 0..255
  int mh = w & 1;                         // 0..1
  int n  = nt * 16 + ln;
  int mbase = mh * 4;

  f32x8 acc[4];
#pragma unroll
  for (int t = 0; t < 4; ++t) acc[t] = 0.f;

  const float* wih = W_ih + (size_t)n * 768 + khalf;
  for (int kb = 0; kb < 768; kb += 32)
    lstm_kstep(Xb, wih + kb, kb, half, ln, mbase, acc);

  const float* whh = W_hh + (size_t)n * 1024 + khalf;
  for (int kb = 768; kb < 1792; kb += 32)
    lstm_kstep(Xb, whh + (kb - 768), kb, half, ln, mbase, acc);

#pragma unroll
  for (int t = 0; t < 4; ++t) {
#pragma unroll
    for (int r = 0; r < 8; ++r)
      gates[(size_t)((mbase + t) * 16 + r + 8 * half) * 4096 + n] = acc[t][r];
  }
}

// ---------------------------------------------------------------------------
// Kernel 4: LSTM pointwise gating (PyTorch gate order i,f,g,o) -> h_new
// ---------------------------------------------------------------------------
__global__ void k_lstm_point(const float* __restrict__ gates,
                             const float* __restrict__ b_ih, const float* __restrict__ b_hh,
                             const float* __restrict__ c_in, float* __restrict__ h_out) {
  int idx = blockIdx.x * blockDim.x + threadIdx.x;
  if (idx >= 128 * 1024) return;
  int b = idx >> 10, j = idx & 1023;
  const float* g = gates + (size_t)b * 4096;
  float gi = g[j]        + b_ih[j]        + b_hh[j];
  float gf = g[1024 + j] + b_ih[1024 + j] + b_hh[1024 + j];
  float gg = g[2048 + j] + b_ih[2048 + j] + b_hh[2048 + j];
  float go = g[3072 + j] + b_ih[3072 + j] + b_hh[3072 + j];
  float c = sigmoidf_(gf) * c_in[idx] + sigmoidf_(gi) * tanhf(gg);
  h_out[idx] = sigmoidf_(go) * tanhf(c);
}

// ---------------------------------------------------------------------------
// Kernel 5: qmb[128,128] = h @ q_w^T + q_b + m_b + b2   (M=N=128, K=1024)
// ---------------------------------------------------------------------------
__global__ __launch_bounds__(256) void k_query(
    const float* __restrict__ h, const float* __restrict__ q_w,
    const float* __restrict__ q_b, const float* __restrict__ m_b,
    const float* __restrict__ b2, float* __restrict__ qmb) {
  int lane = threadIdx.x & 31, wave = threadIdx.x >> 5;
  int half = lane >> 4, ln = lane & 15, khalf = half * 16;
  int id = blockIdx.x * 8 + wave;           // 64 tiles
  int mt = id >> 3, nt = id & 7;
  int n = nt * 16 + ln;

  f32x8 acc = 0.f;
  const float* arow = h   + (size_t)(mt * 16 + ln) * 1024;
  const float* brow = q_w + (size_t)n * 1024 + khalf;
  for (int kb = 0; kb < 1024; kb += 32) {
    bf16x16 bf = load_b_f32cvt(brow + kb);
    bf16x16 af = load_a_f32cvt(arow, kb, half);
    acc = WMMA_BF16(af, bf, acc);
  }
  float extra = q_b[n] + m_b[n] + b2[n];
#pragma unroll
  for (int r = 0; r < 8; ++r)
    qmb[(size_t)(mt * 16 + r + 8 * half) * 128 + n] = acc[r] + extra;
}

// ---------------------------------------------------------------------------
// Kernel 6: main attention. One workgroup (8 waves) per batch row.
//  - m_w staged in LDS as bf16 [128][512+8] (pad spreads banks) : 133 KB
//  - each wave processes 2 adjacent S-tiles per pass (B frags reused 2x)
//  - energy = tanh(pm + loc + qmb), logits via o_w dot + shfl reduce
//  - block softmax (PAD mask), then context GEMV over encoded_text
// ---------------------------------------------------------------------------
__global__ __launch_bounds__(256) void k_attn(
    const float* __restrict__ enc,      // [128,512,512]
    const float* __restrict__ prev_aw,  // [128,512]
    const float* __restrict__ cum_aw,   // [128,512]
    const float* __restrict__ m_w,      // [128,512]
    const float* __restrict__ o_w,      // [128]
    const int*   __restrict__ text,     // [128,512]
    const float* __restrict__ qmb,      // [128,128]
    const __bf16* __restrict__ w2b,     // [128,64]
    float* __restrict__ out) {          // [128,512]
  __shared__ __bf16 mwlds[128][520];    // padded stride: avoid LDS bank camping
  __shared__ float  attn[2][560];       // loc_in with 15-left pad, zero halo
  __shared__ float  qml[128];
  __shared__ float  owl[128];
  __shared__ float  logits[512];
  __shared__ float  wts[512];
  __shared__ float  red[40];

  int b   = blockIdx.x;
  int tid = threadIdx.x;

  for (int idx = tid; idx < 128 * 512; idx += 256) {
    int a = idx >> 9, k = idx & 511;
    mwlds[a][k] = (__bf16)m_w[(size_t)a * 512 + k];
  }
  for (int i = tid; i < 2 * 560; i += 256) (&attn[0][0])[i] = 0.f;
  __syncthreads();
  for (int s = tid; s < 512; s += 256) {
    attn[0][15 + s] = cum_aw[(size_t)b * 512 + s];
    attn[1][15 + s] = prev_aw[(size_t)b * 512 + s];
  }
  if (tid < 128) { qml[tid] = qmb[(size_t)b * 128 + tid]; owl[tid] = o_w[tid]; }
  __syncthreads();

  int lane = tid & 31, wave = tid >> 5;
  int half = lane >> 4, ln = lane & 15, khalf = half * 16;
  const float* encb = enc + (size_t)b * 512 * 512;

  // Each wave: 2 passes x 2 adjacent S-tiles (32 rows per pass).
  for (int g = 0; g < 2; ++g) {
    int s0 = wave * 32 + g * 256;                  // first of two 16-row tiles
    f32x8 acc[2][8];
#pragma unroll
    for (int u = 0; u < 2; ++u)
#pragma unroll
      for (int t = 0; t < 8; ++t) acc[u][t] = 0.f;

    const float* arow0 = encb + (size_t)(s0 + ln) * 512;
    const float* arow1 = encb + (size_t)(s0 + 16 + ln) * 512;
    // prefetch this wave's rows for the next pass
    __builtin_prefetch(arow0 + 256 * 512, 0, 0);
    __builtin_prefetch(arow1 + 256 * 512, 0, 0);

    // pm = enc[s0:s0+32,:] @ m_w^T : 16 K-steps of bf16 WMMA
    for (int kb = 0; kb < 512; kb += 32) {
      bf16x16 af0 = load_a_f32cvt(arow0, kb, half);
      bf16x16 af1 = load_a_f32cvt(arow1, kb, half);
#pragma unroll
      for (int t = 0; t < 8; ++t) {
        const __bf16* mp = &mwlds[t * 16 + ln][kb + khalf];
        bf16x16 bf;
#pragma unroll
        for (int j = 0; j < 16; ++j) bf[j] = mp[j];
        acc[0][t] = WMMA_BF16(af0, bf, acc[0][t]);
        acc[1][t] = WMMA_BF16(af1, bf, acc[1][t]);
      }
    }

    // fused location conv as im2col GEMM, K=64 (2 WMMA K-steps)
    for (int kb = 0; kb < 64; kb += 32) {
      bf16x16 af0, af1;
#pragma unroll
      for (int j = 0; j < 16; ++j) {
        int jj = kb + half * 16 + j;
        int c = jj >> 5, kk = jj & 31;
        af0[j] = (__bf16)attn[c][s0 + ln + kk];
        af1[j] = (__bf16)attn[c][s0 + 16 + ln + kk];
      }
#pragma unroll
      for (int t = 0; t < 8; ++t) {
        const __bf16* wp = w2b + (size_t)(t * 16 + ln) * 64 + kb + khalf;
        bf16x16 bf;
#pragma unroll
        for (int j = 0; j < 16; ++j) bf[j] = wp[j];
        acc[0][t] = WMMA_BF16(af0, bf, acc[0][t]);
        acc[1][t] = WMMA_BF16(af1, bf, acc[1][t]);
      }
    }

    // energy -> logits (tanh, o_w dot, cross-lane reduce within 16-lane half)
#pragma unroll
    for (int u = 0; u < 2; ++u) {
#pragma unroll
      for (int r = 0; r < 8; ++r) {
        float partial = 0.f;
#pragma unroll
        for (int t = 0; t < 8; ++t) {
          int a = t * 16 + ln;
          partial += owl[a] * tanhf(acc[u][t][r] + qml[a]);
        }
        for (int off = 1; off < 16; off <<= 1)
          partial += __shfl_xor(partial, off, 32);
        if (ln == 0) logits[s0 + u * 16 + r + 8 * half] = partial;
      }
    }
  }
  __syncthreads();

  // softmax over S=512 with PAD mask (o_b is a uniform shift: softmax-invariant)
  float l2[2];
  float lmax = -1e30f;
#pragma unroll
  for (int i = 0; i < 2; ++i) {
    int s = tid + i * 256;
    float l = logits[s] + ((text[(size_t)b * 512 + s] == 0) ? -1e9f : 0.f);
    l2[i] = l;
    lmax = fmaxf(lmax, l);
  }
  for (int off = 1; off < 32; off <<= 1) lmax = fmaxf(lmax, __shfl_xor(lmax, off, 32));
  if (lane == 0) red[wave] = lmax;
  __syncthreads();
  if (tid == 0) {
    float m = red[0];
    for (int w = 1; w < 8; ++w) m = fmaxf(m, red[w]);
    red[32] = m;
  }
  __syncthreads();
  float gmax = red[32];
  float psum = 0.f;
#pragma unroll
  for (int i = 0; i < 2; ++i) {
    int s = tid + i * 256;
    float e = expf(l2[i] - gmax);
    wts[s] = e;
    psum += e;
  }
  for (int off = 1; off < 32; off <<= 1) psum += __shfl_xor(psum, off, 32);
  if (lane == 0) red[8 + wave] = psum;
  __syncthreads();
  if (tid == 0) {
    float s = 0.f;
    for (int w = 0; w < 8; ++w) s += red[8 + w];
    red[33] = s;
  }
  __syncthreads();
  float inv = 1.f / red[33];

  // context[b,e] = sum_s w[s] * enc[b,s,e]  (coalesced; enc tile is L2-hot)
#pragma unroll
  for (int i = 0; i < 2; ++i) {
    int e = tid + i * 256;
    float accc = 0.f;
    for (int s = 0; s < 512; ++s) accc += wts[s] * encb[(size_t)s * 512 + e];
    out[(size_t)b * 512 + e] = accc * inv;
  }
}

// ---------------------------------------------------------------------------
// Host launcher
// ---------------------------------------------------------------------------
extern "C" void kernel_launch(void* const* d_in, const int* in_sizes, int n_in,
                              void* d_out, int out_size, void* d_ws, size_t ws_size,
                              hipStream_t stream) {
  const float* prenet   = (const float*)d_in[0];
  const float* enc      = (const float*)d_in[1];
  const float* prev_ctx = (const float*)d_in[2];
  const float* att_h    = (const float*)d_in[3];
  const float* att_c    = (const float*)d_in[4];
  const float* prev_aw  = (const float*)d_in[5];
  const float* cum_aw   = (const float*)d_in[6];
  const float* W_ih     = (const float*)d_in[7];
  const float* W_hh     = (const float*)d_in[8];
  const float* b_ih     = (const float*)d_in[9];
  const float* b_hh     = (const float*)d_in[10];
  const float* conv_w   = (const float*)d_in[11];
  const float* conv_b   = (const float*)d_in[12];
  const float* loc_w    = (const float*)d_in[13];
  const float* loc_b    = (const float*)d_in[14];
  const float* q_w      = (const float*)d_in[15];
  const float* q_b      = (const float*)d_in[16];
  const float* m_w      = (const float*)d_in[17];
  const float* m_b      = (const float*)d_in[18];
  const float* o_w      = (const float*)d_in[19];
  const int*   text     = (const int*)d_in[21];
  float* out = (float*)d_out;

  char* ws = (char*)d_ws;
  float*  gates = (float*)(ws + 0);                 // 128*4096*4 = 2 MB
  float*  h_ws  = (float*)(ws + 2097152);           // 128*1024*4 = 512 KB
  float*  qmb   = (float*)(ws + 2621440);           // 128*128*4  = 64 KB
  float*  b2    = (float*)(ws + 2686976);           // 128*4
  __bf16* w2b   = (__bf16*)(ws + 2687488);          // 128*64*2   = 16 KB
  __bf16* Xb    = (__bf16*)(ws + 2703872);          // 128*1792*2 = 448 KB

  k_prep<<<1, 128, 0, stream>>>(conv_w, conv_b, loc_w, loc_b, w2b, b2);
  k_pack<<<dim3(7, 128), 256, 0, stream>>>(prenet, prev_ctx, att_h, Xb);
  k_lstm_gemm<<<64, 256, 0, stream>>>(Xb, W_ih, W_hh, gates);
  k_lstm_point<<<512, 256, 0, stream>>>(gates, b_ih, b_hh, att_c, h_ws);
  k_query<<<8, 256, 0, stream>>>(h_ws, q_w, q_b, m_b, b2, qmb);
  k_attn<<<128, 256, 0, stream>>>(enc, prev_aw, cum_aw, m_w, o_w, text, qmb, w2b, out);
}
